// Attention_36833639530707
// MI455X (gfx1250) — matrix operand order repaired and checked by reading, not compiled, not run on hardware
//
#include <hip/hip_runtime.h>
#include <hip/hip_bf16.h>

#define B_ 32
#define S_ 2048
#define H_ 1024
#define AS_STRIDE 1032   // 1024 bf16 + 8 pad (16B) -> kills LDS bank conflicts

typedef __attribute__((ext_vector_type(16))) __bf16 v16bf;
typedef __attribute__((ext_vector_type(8)))  __bf16 v8bf;
typedef __attribute__((ext_vector_type(8)))  float  v8f;

__device__ __forceinline__ unsigned short f2bfu(float f) {
  union { float f; unsigned u; } x; x.f = f;
  return (unsigned short)((x.u + 0x7FFFu + ((x.u >> 16) & 1u)) >> 16);  // RNE
}

// Hardware V_TANH_F32 on gfx1250 if the toolchain exposes it; OCML fallback.
#if __has_builtin(__builtin_amdgcn_tanhf)
#define TANHF(x) __builtin_amdgcn_tanhf(x)
#else
#define TANHF(x) tanhf(x)
#endif

// ---- pass 0a: We (= W_w[:, H:2H]) -> bf16, row-major [o][k] ----
__global__ void att_wt_bf16(const float* __restrict__ W_w, unsigned short* __restrict__ Wt) {
  int idx = blockIdx.x * 256 + threadIdx.x;          // 0 .. 1048575
  int o = idx >> 10, k = idx & 1023;
  Wt[idx] = f2bfu(W_w[(size_t)o * (2 * H_) + H_ + k]);
}

// ---- pass 0b: hb[b,o] = hidden[b,:] . W_w[o,0:H] + W_b[o] ----
__global__ void att_hb(const float* __restrict__ hidden, const float* __restrict__ W_w,
                       const float* __restrict__ W_b, float* __restrict__ hb) {
  int idx = blockIdx.x * 256 + threadIdx.x;          // 0 .. 32767
  int b = idx >> 10, o = idx & 1023;
  const float* hr = hidden + (size_t)b * H_;
  const float* wr = W_w + (size_t)o * (2 * H_);
  float s = W_b[o];
#pragma unroll 4
  for (int h = 0; h < H_; ++h) s += hr[h] * wr[h];
  hb[idx] = s;
}

// ---- pass 1: fused GEMM + tanh + v-dot -> scores[b,s] (pre-softmax) ----
// grid = 512 (each block = 128 consecutive (b,s) rows, 128 | S so b is uniform)
// block = 256 threads = 8 waves; wave (mq = w&3, q = w>>2) owns M-subtiles
// {2mq, 2mq+1} and N-subtiles {2q, 2q+1} of each 64-column chunk.
__global__ void __launch_bounds__(256)
att_scores(const float* __restrict__ enc, const unsigned short* __restrict__ WtU,
           const float* __restrict__ hb, const float* __restrict__ vw,
           float* __restrict__ scores) {
  __shared__ __align__(16) unsigned short As[128 * AS_STRIDE];  // 258 KB of the 320 KB WGP LDS
  __shared__ float scoreLDS[128];

  const int tid  = threadIdx.x;
  const int w    = tid >> 5;
  const int lane = tid & 31;
  const int r    = lane & 15;
  const int half = lane >> 4;
  const int mq   = w & 3;
  const int q    = w >> 2;
  const int r0   = blockIdx.x * 128;
  const int b    = blockIdx.x >> 4;

  if (tid < 128) scoreLDS[tid] = 0.f;

  // Stage full-K A tile: 128 x 1024 f32 -> bf16 in LDS (enc read exactly once)
  for (int row = 0; row < 128; ++row) {
    const float4 v = ((const float4*)(enc + (size_t)(r0 + row) * H_))[tid];
    unsigned lo = (unsigned)f2bfu(v.x) | ((unsigned)f2bfu(v.y) << 16);
    unsigned hi = (unsigned)f2bfu(v.z) | ((unsigned)f2bfu(v.w) << 16);
    *(uint2*)&As[row * AS_STRIDE + tid * 4] = make_uint2(lo, hi);
  }
  __syncthreads();

  const __bf16* Wt = (const __bf16*)WtU;
  float sacc[2][8];
#pragma unroll
  for (int mi = 0; mi < 2; ++mi)
#pragma unroll
    for (int v = 0; v < 8; ++v) sacc[mi][v] = 0.f;

  const int m0row = ((mq * 2 + 0) * 16 + r) * AS_STRIDE;
  const int m1row = ((mq * 2 + 1) * 16 + r) * AS_STRIDE;

#pragma unroll 1
  for (int chunk = 0; chunk < 16; ++chunk) {
    const int n0 = chunk * 64 + (q * 2) * 16;   // this wave's two o-column tiles
    const int n1 = n0 + 16;
    // B fragment: lane holds column n0+r; VGPRs 0-3: K = half*8+[0..7], VGPRs 4-7: +16
    const __bf16* bp0 = Wt + (size_t)(n0 + r) * H_ + half * 8;
    const __bf16* bp1 = Wt + (size_t)(n1 + r) * H_ + half * 8;

    // Prefetch epilogue scalars before the K loop so their L2 latency hides
    // under the 32 WMMA iterations.
    const int oc0 = n0 + r, oc1 = n1 + r;
    const float hb0 = hb[b * H_ + oc0], vw0 = vw[oc0];
    const float hb1 = hb[b * H_ + oc1], vw1 = vw[oc1];

    v8f c00 = {}, c01 = {}, c10 = {}, c11 = {};

#pragma unroll 4
    for (int k0 = 0; k0 < H_; k0 += 32) {
      union { v16bf v; v8bf h[2]; } a0, a1, b0, b1;
      // A fragment: lane holds row m*16+r; VGPRs 0-3: K = half*8+[0..7], VGPRs 4-7: +16
      a0.h[0] = *(const v8bf*)&As[m0row + k0 + half * 8];
      a0.h[1] = *(const v8bf*)&As[m0row + k0 + 16 + half * 8];
      a1.h[0] = *(const v8bf*)&As[m1row + k0 + half * 8];
      a1.h[1] = *(const v8bf*)&As[m1row + k0 + 16 + half * 8];
      b0.h[0] = *(const v8bf*)(bp0 + k0);
      b0.h[1] = *(const v8bf*)(bp0 + k0 + 16);
      b1.h[0] = *(const v8bf*)(bp1 + k0);
      b1.h[1] = *(const v8bf*)(bp1 + k0 + 16);
      c00 = __builtin_amdgcn_wmma_f32_16x16x32_bf16(false, a0.v, false, b0.v, (short)0, c00, false, false);
      c01 = __builtin_amdgcn_wmma_f32_16x16x32_bf16(false, a0.v, false, b1.v, (short)0, c01, false, false);
      c10 = __builtin_amdgcn_wmma_f32_16x16x32_bf16(false, a1.v, false, b0.v, (short)0, c10, false, false);
      c11 = __builtin_amdgcn_wmma_f32_16x16x32_bf16(false, a1.v, false, b1.v, (short)0, c11, false, false);
    }

    // epilogue: C/D layout lane = column (n+r), VGPR v = row v + 8*half
#pragma unroll
    for (int v = 0; v < 8; ++v) {
      sacc[0][v] += TANHF(c00[v] + hb0) * vw0;
      sacc[0][v] += TANHF(c01[v] + hb1) * vw1;
      sacc[1][v] += TANHF(c10[v] + hb0) * vw0;
      sacc[1][v] += TANHF(c11[v] + hb1) * vw1;
    }
  }

  // column-reduce within each 16-lane half; row = (mq*2+mi)*16 + half*8 + v
#pragma unroll
  for (int mi = 0; mi < 2; ++mi) {
#pragma unroll
    for (int v = 0; v < 8; ++v) {
      float val = sacc[mi][v];
#pragma unroll
      for (int off = 1; off < 16; off <<= 1) val += __shfl_xor(val, off, 32);
      if (r == 0) atomicAdd(&scoreLDS[(mq * 2 + mi) * 16 + half * 8 + v], val);
    }
  }
  __syncthreads();
  if (tid < 128) scores[r0 + tid] = scoreLDS[tid];
}

// ---- pass 2: softmax over s (shift-invariant: v_b dropped) ----
__global__ void att_softmax(const float* __restrict__ scores, float* __restrict__ weights) {
  const int b = blockIdx.x, tid = threadIdx.x;   // 256 threads
  __shared__ float sh[S_];
  __shared__ float red[256];
  const float* sr = scores + (size_t)b * S_;
  float m = -3.4e38f;
  for (int i = tid; i < S_; i += 256) { float v = sr[i]; sh[i] = v; m = fmaxf(m, v); }
  red[tid] = m; __syncthreads();
  for (int off = 128; off > 0; off >>= 1) { if (tid < off) red[tid] = fmaxf(red[tid], red[tid + off]); __syncthreads(); }
  const float mx = red[0]; __syncthreads();
  float s = 0.f;
  for (int i = tid; i < S_; i += 256) { float e = expf(sh[i] - mx); sh[i] = e; s += e; }
  red[tid] = s; __syncthreads();
  for (int off = 128; off > 0; off >>= 1) { if (tid < off) red[tid] += red[tid + off]; __syncthreads(); }
  const float inv = 1.f / red[0];
  for (int i = tid; i < S_; i += 256) weights[(size_t)b * S_ + i] = sh[i] * inv;
}

// ---- pass 3: context[b,h] = sum_s w[b,s] * enc[b,s,h] ----
__global__ void att_context(const float* __restrict__ enc, const float* __restrict__ weights,
                            float* __restrict__ context) {
  const int b = blockIdx.x, hc = blockIdx.y, tid = threadIdx.x;  // 128 threads, 8 h-chunks
  __shared__ float w[S_];
  for (int i = tid; i < S_; i += 128) w[i] = weights[(size_t)b * S_ + i];
  __syncthreads();
  const int h = hc * 128 + tid;
  const float* e = enc + ((size_t)b * S_) * H_ + h;
  float acc = 0.f;
#pragma unroll 4
  for (int s = 0; s < S_; ++s) acc += w[s] * e[(size_t)s * H_];
  context[(size_t)b * H_ + h] = acc;
}

extern "C" void kernel_launch(void* const* d_in, const int* in_sizes, int n_in,
                              void* d_out, int out_size, void* d_ws, size_t ws_size,
                              hipStream_t stream) {
  (void)in_sizes; (void)n_in; (void)out_size; (void)ws_size;
  const float* hidden = (const float*)d_in[0];
  const float* enc    = (const float*)d_in[1];
  const float* W_w    = (const float*)d_in[2];
  const float* W_b    = (const float*)d_in[3];
  const float* v_w    = (const float*)d_in[4];
  // d_in[5] (v_b) intentionally unused: softmax is invariant to a uniform shift.

  float* out     = (float*)d_out;
  float* context = out;              // B*H = 32768 floats
  float* weights = out + B_ * H_;    // B*S = 65536 floats

  char* ws = (char*)d_ws;
  float*          scores = (float*)ws;                      // 256 KB
  float*          hb     = (float*)(ws + 262144);           // 128 KB
  unsigned short* Wt     = (unsigned short*)(ws + 393216);  // 2 MB bf16

  att_wt_bf16<<<dim3((H_ * H_) / 256), dim3(256), 0, stream>>>(W_w, Wt);
  att_hb<<<dim3((B_ * H_) / 256), dim3(256), 0, stream>>>(hidden, W_w, W_b, hb);
  att_scores<<<dim3((B_ * S_) / 128), dim3(256), 0, stream>>>(enc, Wt, hb, v_w, scores);
  att_softmax<<<dim3(B_), dim3(256), 0, stream>>>(scores, weights);
  att_context<<<dim3(B_, H_ / 128), dim3(128), 0, stream>>>(enc, weights, context);
}
